// MultiHeadSelfAttention_19215683682340
// MI455X (gfx1250) — matrix operand order repaired and checked by reading, not compiled, not run on hardware
//
#include <hip/hip_runtime.h>
#include <hip/hip_bf16.h>
#include <math.h>

// ---------------------------------------------------------------------------
// MHA for MI455X (gfx1250, wave32): bf16 WMMA everywhere, f32 accumulate.
// Stages: convert/pack -> fused QKV GEMM (64x64/wave) ->
//         flash attention (async-to-LDS K/V staging shared by 4 waves) ->
//         out GEMM (64x64/wave).
// ---------------------------------------------------------------------------

typedef __attribute__((ext_vector_type(16))) __bf16 v16bf;
typedef __attribute__((ext_vector_type(8)))  __bf16 v8bf;
typedef __attribute__((ext_vector_type(8)))  float  v8f;

constexpr int   Bb = 4, Nn = 2048, Cc = 1024, Hh = 16, Dd = 64;
constexpr int   BN = Bb * Nn;                  // 8192 rows
constexpr float QSCALE = 0.125f;               // 1/sqrt(D)
constexpr float LOG2E  = 1.44269504088896340736f;

// D = A*B + C   (16x16 f32 out, A 16x32 bf16, B 32x16 bf16)
__device__ __forceinline__ v8f wmma_bf16(v16bf a, v16bf b, v8f c) {
  return __builtin_amdgcn_wmma_f32_16x16x32_bf16(
      /*neg_a=*/false, a, /*neg_b=*/false, b,
      /*c_mod=*/(short)0, c, /*reuse_a=*/false, /*reuse_b=*/false);
}

// A-operand loader: row-major [16 x 32] bf16 tile, row stride `ld` elements.
// ISA layout: lanes 0-15 hold row m, K = {kb..kb+7, 16+kb..16+kb+7}, kb = 8*(lane>=16).
__device__ __forceinline__ v16bf load_a16(const __bf16* base, int ld, int lane) {
  const int m  = lane & 15;
  const int kb = (lane >> 4) << 3;             // 0 or 8
  const __bf16* p = base + m * ld + kb;
  v8bf lo = *(const v8bf*)(p);                 // K kb .. kb+7
  v8bf hi = *(const v8bf*)(p + 16);            // K 16+kb .. 16+kb+7
  v16bf a;
#pragma unroll
  for (int j = 0; j < 8; ++j) { a[j] = lo[j]; a[j + 8] = hi[j]; }
  return a;
}

// B-operand loader from column-major-packed storage Bt[n][k]: lane holds
// column n = lane&15, K = {kb..kb+7, 16+kb..16+kb+7}, kb = 8*(lane>=16).
__device__ __forceinline__ v16bf load_bt(const __bf16* base, int ld, int lane) {
  const int n  = lane & 15;
  const int kb = (lane >> 4) << 3;             // 0 or 8
  const __bf16* p = base + n * ld + kb;
  v8bf lo = *(const v8bf*)(p);
  v8bf hi = *(const v8bf*)(p + 16);
  v16bf b;
#pragma unroll
  for (int j = 0; j < 8; ++j) { b[j] = lo[j]; b[j + 8] = hi[j]; }
  return b;
}

// CDNA5 async copy: one 16-byte chunk global -> LDS, tracked by ASYNCcnt.
__device__ __forceinline__ void async_g2l_b128(const __bf16* lds_dst,
                                               const __bf16* gsrc) {
  unsigned int       l = (unsigned int)(size_t)lds_dst;     // low 32b = LDS addr
  unsigned long long g = (unsigned long long)(size_t)gsrc;
  asm volatile("global_load_async_to_lds_b128 %0, %1, off"
               :: "v"(l), "v"(g) : "memory");
}

// 16-lane-group reductions (masks 1..8 stay inside each half of wave32)
__device__ __forceinline__ float rmax16(float v) {
#pragma unroll
  for (int m = 1; m <= 8; m <<= 1) v = fmaxf(v, __shfl_xor(v, m, 32));
  return v;
}
__device__ __forceinline__ float rsum16(float v) {
#pragma unroll
  for (int m = 1; m <= 8; m <<= 1) v += __shfl_xor(v, m, 32);
  return v;
}

// ---------------------------------------------------------------------------
// Stage 1: conversions
// ---------------------------------------------------------------------------
__global__ __launch_bounds__(256) void mha_cvt_bf16(const float* __restrict__ in,
                                                    __bf16* __restrict__ out, int n) {
  int i = blockIdx.x * 256 + threadIdx.x;
  if (i < n) out[i] = (__bf16)in[i];
}

// W[c][d] (row-major [C,C]) -> WT[d][c] bf16, so GEMM B-operand is contiguous-k.
__global__ __launch_bounds__(256) void mha_cvt_wT(const float* __restrict__ W,
                                                  __bf16* __restrict__ WT) {
  int i = blockIdx.x * 256 + threadIdx.x;     // i = c*C + d
  int c = i >> 10;                             // /C
  int d = i & (Cc - 1);
  WT[d * Cc + c] = (__bf16)W[i];
}

// ---------------------------------------------------------------------------
// Stage 2: fused QKV projection.  One wave per 64x64 output block
// (16 WMMA accumulators; 4 A-loads + 4 B-loads feed 16 WMMAs per k-step).
//   Q -> Qs[B,H,N,D] (pre-scaled by 1/sqrt(D))
//   K -> Kb[B,H,N,D]            (QK^T B-operand contiguous)
//   V -> Vt[B,H,D,N]            (PV   B-operand contiguous)
// ---------------------------------------------------------------------------
__global__ __launch_bounds__(128) void mha_qkv_gemm(
    const __bf16* __restrict__ xb,                        // [BN, C]
    const __bf16* __restrict__ WqT, const __bf16* __restrict__ WkT,
    const __bf16* __restrict__ WvT,
    const float* __restrict__ bq, const float* __restrict__ bk,
    const float* __restrict__ bv,
    __bf16* __restrict__ Qs, __bf16* __restrict__ Kb, __bf16* __restrict__ Vt) {
  const int lane = threadIdx.x & 31;
  const int wave = threadIdx.x >> 5;
  constexpr int colBlocks = Cc / 64;                      // 16
  constexpr int perMat    = (BN / 64) * colBlocks;        // 2048
  int tile = blockIdx.x * 4 + wave;
  int mat  = tile / perMat;                               // 0=Q 1=K 2=V
  int t    = tile - mat * perMat;
  int rb   = t / colBlocks;
  int cb   = t - rb * colBlocks;

  const __bf16* WT   = (mat == 0) ? WqT : (mat == 1) ? WkT : WvT;
  const float*  bias = (mat == 0) ? bq  : (mat == 1) ? bk  : bv;

  const __bf16* aBase = xb + (size_t)rb * 64 * Cc;
  const __bf16* bBase = WT + (size_t)cb * 64 * Cc;

  v8f acc[4][4];
#pragma unroll
  for (int i = 0; i < 4; ++i)
#pragma unroll
    for (int j = 0; j < 4; ++j)
      acc[i][j] = (v8f){0.f, 0.f, 0.f, 0.f, 0.f, 0.f, 0.f, 0.f};

  for (int k = 0; k < Cc; k += 32) {
    v16bf a[4];
#pragma unroll
    for (int i = 0; i < 4; ++i) a[i] = load_a16(aBase + i * 16 * Cc + k, Cc, lane);
#pragma unroll
    for (int j = 0; j < 4; ++j) {
      v16bf b = load_bt(bBase + j * 16 * Cc + k, Cc, lane);
#pragma unroll
      for (int i = 0; i < 4; ++i) acc[i][j] = wmma_bf16(a[i], b, acc[i][j]);
    }
  }

  const int cl = lane & 15;
  const int hi = lane >> 4;
#pragma unroll
  for (int j = 0; j < 4; ++j) {
    const int d  = cb * 64 + j * 16 + cl;
    const float bia = bias[d];
    const int h  = d >> 6;                                 // D = 64
    const int dd = d & 63;
#pragma unroll
    for (int i = 0; i < 4; ++i) {
#pragma unroll
      for (int r = 0; r < 8; ++r) {
        int gn = rb * 64 + i * 16 + r + (hi << 3);         // global row in [BN)
        int b_ = gn >> 11;                                 // /N
        int n_ = gn & (Nn - 1);
        float y = acc[i][j][r] + bia;
        if (mat == 0)
          Qs[(((b_ * Hh + h) * Nn) + n_) * Dd + dd] = (__bf16)(y * QSCALE);
        else if (mat == 1)
          Kb[(((b_ * Hh + h) * Nn) + n_) * Dd + dd] = (__bf16)y;
        else
          Vt[(((b_ * Hh + h) * Dd) + dd) * Nn + n_] = (__bf16)y;
      }
    }
  }
}

// ---------------------------------------------------------------------------
// Stage 3: flash attention.  grid = (N/128, B*H); 4 waves/block share one
// (b,h): each m-block of 32 keys is staged K[32x64] / V[64x32] into LDS once
// via global_load_async_to_lds_b128 and consumed by all 4 waves.  Each wave
// owns 32 Q rows (2 row-tiles).
// ---------------------------------------------------------------------------
__global__ __launch_bounds__(128) void mha_flash_attn(
    const __bf16* __restrict__ Qs, const __bf16* __restrict__ Kb,
    const __bf16* __restrict__ Vt, __bf16* __restrict__ Ob) {
  __shared__ __attribute__((aligned(32))) __bf16 kls[32 * 64];      // 4 KB
  __shared__ __attribute__((aligned(32))) __bf16 vls[64 * 32];      // 4 KB
  __shared__ __attribute__((aligned(32))) __bf16 pls[4][2][16 * 32];// 8 KB

  const int tid  = threadIdx.x;
  const int lane = tid & 31;
  const int wave = tid >> 5;
  const int bh   = blockIdx.y;                             // b*H + h
  const int qb   = blockIdx.x * 128;                       // Q rows of block

  const __bf16* kbase = Kb + (size_t)bh * Nn * Dd;
  const __bf16* vbase = Vt + (size_t)bh * Dd * Nn;
  const __bf16* q     = Qs + (size_t)bh * Nn * Dd + (size_t)(qb + wave * 32) * Dd;

  v16bf aq[2][2];
#pragma unroll
  for (int rt = 0; rt < 2; ++rt) {
    aq[rt][0] = load_a16(q + rt * 16 * Dd, Dd, lane);      // d = 0..31
    aq[rt][1] = load_a16(q + rt * 16 * Dd + 32, Dd, lane); // d = 32..63
  }

  v8f acc[2][4];
  float rm[2][8], rl[2][8];
#pragma unroll
  for (int rt = 0; rt < 2; ++rt) {
#pragma unroll
    for (int t = 0; t < 4; ++t)
      acc[rt][t] = (v8f){0.f, 0.f, 0.f, 0.f, 0.f, 0.f, 0.f, 0.f};
#pragma unroll
    for (int r = 0; r < 8; ++r) { rm[rt][r] = -INFINITY; rl[rt][r] = 0.f; }
  }

  const v8f zero = {0.f, 0.f, 0.f, 0.f, 0.f, 0.f, 0.f, 0.f};
  const int hi = lane >> 4;
  const int cl = lane & 15;

  for (int m0 = 0; m0 < Nn; m0 += 32) {
    __syncthreads();   // prior iteration's LDS readers are done

    // ---- cooperative async staging: K block 32x64 (256 x 16B chunks),
    //      V block 64x32 (256 x 16B chunks); 2+2 chunks per thread.
#pragma unroll
    for (int c = 0; c < 2; ++c) {
      int idx = tid + c * 128;
      int krow = idx >> 3, kc = (idx & 7) * 8;             // K: 8 chunks/row
      async_g2l_b128(kls + krow * 64 + kc,
                     kbase + (size_t)(m0 + krow) * Dd + kc);
      int vrow = idx >> 2, vc = (idx & 3) * 8;             // V: 4 chunks/row
      async_g2l_b128(vls + vrow * 32 + vc,
                     vbase + (size_t)vrow * Nn + m0 + vc);
    }
    // prefetch next K/V block toward L2 while this one lands
    if (m0 + 32 < Nn) {
      __builtin_prefetch(kbase + (size_t)(m0 + 32) * Dd + (size_t)lane * 64, 0, 0);
      __builtin_prefetch(vbase + (size_t)lane * Nn + m0 + 32, 0, 0);
      __builtin_prefetch(vbase + (size_t)(32 + lane) * Nn + m0 + 32, 0, 0);
    }
    asm volatile("s_wait_asynccnt 0x0" ::: "memory");
    __syncthreads();   // staged K/V visible to all waves

    // ---- S = Q K^T from LDS (2 row-tiles x 2 key-tiles, K-dim 64 -> 2 WMMA)
    v16bf ap[2];
#pragma unroll
    for (int rt = 0; rt < 2; ++rt) {
      v8f s0 = wmma_bf16(aq[rt][0], load_bt(kls, 64, lane), zero);
      s0     = wmma_bf16(aq[rt][1], load_bt(kls + 32, 64, lane), s0);
      v8f s1 = wmma_bf16(aq[rt][0], load_bt(kls + 16 * 64, 64, lane), zero);
      s1     = wmma_bf16(aq[rt][1], load_bt(kls + 16 * 64 + 32, 64, lane), s1);

      // online softmax per output row (r + 8*hi)
      float alpha[8];
#pragma unroll
      for (int r = 0; r < 8; ++r) {
        float mx   = rmax16(fmaxf(s0[r], s1[r]));
        float mnew = fmaxf(rm[rt][r], mx);
        float p0 = exp2f((s0[r] - mnew) * LOG2E);
        float p1 = exp2f((s1[r] - mnew) * LOG2E);
        alpha[r] = exp2f((rm[rt][r] - mnew) * LOG2E);
        rm[rt][r] = mnew;
        rl[rt][r] = rl[rt][r] * alpha[r] + rsum16(p0 + p1);
        s0[r] = p0; s1[r] = p1;
      }
#pragma unroll
      for (int t = 0; t < 4; ++t)
#pragma unroll
        for (int r = 0; r < 8; ++r) acc[rt][t][r] *= alpha[r];

      // P (C-layout f32) -> per-wave LDS -> A-layout bf16 [16 x 32]
      __bf16* pw = pls[wave][rt];
#pragma unroll
      for (int r = 0; r < 8; ++r) {
        pw[(r + 8 * hi) * 32 + cl]      = (__bf16)s0[r];
        pw[(r + 8 * hi) * 32 + 16 + cl] = (__bf16)s1[r];
      }
    }
    asm volatile("s_wait_dscnt 0x0" ::: "memory");         // wave-local DS RAW
    ap[0] = load_a16(pls[wave][0], 32, lane);
    ap[1] = load_a16(pls[wave][1], 32, lane);

    // ---- O += P * V  (V tile shared across both row-tiles)
#pragma unroll
    for (int t = 0; t < 4; ++t) {
      v16bf bv = load_bt(vls + t * 16 * 32, 32, lane);
      acc[0][t] = wmma_bf16(ap[0], bv, acc[0][t]);
      acc[1][t] = wmma_bf16(ap[1], bv, acc[1][t]);
    }
  }

  // ---- normalize + write O tiles into [B, N, C] bf16
  const int b_ = bh >> 4;                                  // /H
  const int h_ = bh & 15;
#pragma unroll
  for (int rt = 0; rt < 2; ++rt) {
#pragma unroll
    for (int r = 0; r < 8; ++r) {
      const float inv = 1.f / rl[rt][r];
      const int n_ = qb + wave * 32 + rt * 16 + r + 8 * hi;
      __bf16* orow = Ob + ((size_t)(b_ * Nn + n_)) * Cc + h_ * 64;
#pragma unroll
      for (int t = 0; t < 4; ++t) orow[t * 16 + cl] = (__bf16)(acc[rt][t][r] * inv);
    }
  }
}

// ---------------------------------------------------------------------------
// Stage 4: output projection  Y = O @ Wo + bo  (f32 out), 64x64 per wave.
// ---------------------------------------------------------------------------
__global__ __launch_bounds__(128) void mha_out_gemm(
    const __bf16* __restrict__ Ob,            // [BN, C] bf16
    const __bf16* __restrict__ WoT,           // [C, C] transposed bf16
    const float* __restrict__ bo, float* __restrict__ y) {
  const int lane = threadIdx.x & 31;
  const int wave = threadIdx.x >> 5;
  constexpr int colBlocks = Cc / 64;                       // 16
  int tile = blockIdx.x * 4 + wave;
  int rb = tile / colBlocks;
  int cb = tile - rb * colBlocks;

  const __bf16* aBase = Ob  + (size_t)rb * 64 * Cc;
  const __bf16* bBase = WoT + (size_t)cb * 64 * Cc;

  v8f acc[4][4];
#pragma unroll
  for (int i = 0; i < 4; ++i)
#pragma unroll
    for (int j = 0; j < 4; ++j)
      acc[i][j] = (v8f){0.f, 0.f, 0.f, 0.f, 0.f, 0.f, 0.f, 0.f};

  for (int k = 0; k < Cc; k += 32) {
    v16bf a[4];
#pragma unroll
    for (int i = 0; i < 4; ++i) a[i] = load_a16(aBase + i * 16 * Cc + k, Cc, lane);
#pragma unroll
    for (int j = 0; j < 4; ++j) {
      v16bf b = load_bt(bBase + j * 16 * Cc + k, Cc, lane);
#pragma unroll
      for (int i = 0; i < 4; ++i) acc[i][j] = wmma_bf16(a[i], b, acc[i][j]);
    }
  }

  const int cl = lane & 15;
  const int hi = lane >> 4;
#pragma unroll
  for (int j = 0; j < 4; ++j) {
    const int d = cb * 64 + j * 16 + cl;
    const float bia = bo[d];
#pragma unroll
    for (int i = 0; i < 4; ++i) {
#pragma unroll
      for (int r = 0; r < 8; ++r) {
        int gn = rb * 64 + i * 16 + r + (hi << 3);
        y[(size_t)gn * Cc + d] = acc[i][j][r] + bia;
      }
    }
  }
}

// ---------------------------------------------------------------------------
extern "C" void kernel_launch(void* const* d_in, const int* in_sizes, int n_in,
                              void* d_out, int out_size, void* d_ws, size_t ws_size,
                              hipStream_t stream) {
  const float* x  = (const float*)d_in[0];
  const float* Wq = (const float*)d_in[1];
  const float* bq = (const float*)d_in[2];
  const float* Wk = (const float*)d_in[3];
  const float* bk = (const float*)d_in[4];
  const float* Wv = (const float*)d_in[5];
  const float* bv = (const float*)d_in[6];
  const float* Wo = (const float*)d_in[7];
  const float* bo = (const float*)d_in[8];
  float* out = (float*)d_out;

  char* ws = (char*)d_ws;
  const size_t MB = 1ull << 20;
  __bf16* xb  = (__bf16*)(ws);               // 16 MiB  [BN, C]
  __bf16* WqT = (__bf16*)(ws + 16 * MB);     //  2 MiB
  __bf16* WkT = (__bf16*)(ws + 18 * MB);     //  2 MiB
  __bf16* WvT = (__bf16*)(ws + 20 * MB);     //  2 MiB
  __bf16* WoT = (__bf16*)(ws + 22 * MB);     //  2 MiB
  __bf16* Qs  = (__bf16*)(ws + 24 * MB);     // 16 MiB  [B,H,N,D]
  __bf16* Kbm = (__bf16*)(ws + 40 * MB);     // 16 MiB  [B,H,N,D]
  __bf16* Vt  = (__bf16*)(ws + 56 * MB);     // 16 MiB  [B,H,D,N]
  __bf16* Ob  = (__bf16*)(ws + 72 * MB);     // 16 MiB  [BN, C]   (88 MiB total)

  // Stage 1: pack to bf16 (weights transposed for contiguous B-operand loads)
  mha_cvt_bf16<<<(BN * Cc) / 256, 256, 0, stream>>>(x, xb, BN * Cc);
  mha_cvt_wT<<<(Cc * Cc) / 256, 256, 0, stream>>>(Wq, WqT);
  mha_cvt_wT<<<(Cc * Cc) / 256, 256, 0, stream>>>(Wk, WkT);
  mha_cvt_wT<<<(Cc * Cc) / 256, 256, 0, stream>>>(Wv, WvT);
  mha_cvt_wT<<<(Cc * Cc) / 256, 256, 0, stream>>>(Wo, WoT);

  // Stage 2: fused QKV projection (3 * 128 * 16 blocks of 64x64, 4 waves/block)
  mha_qkv_gemm<<<(3 * (BN / 64) * (Cc / 64)) / 4, 128, 0, stream>>>(
      xb, WqT, WkT, WvT, bq, bk, bv, Qs, Kbm, Vt);

  // Stage 3: flash attention (grid: N/128 q-blocks x B*H heads)
  mha_flash_attn<<<dim3(Nn / 128, Bb * Hh), 128, 0, stream>>>(Qs, Kbm, Vt, Ob);

  // Stage 4: output projection (128 * 16 blocks of 64x64, 4 waves/block)
  mha_out_gemm<<<((BN / 64) * (Cc / 64)) / 4, 128, 0, stream>>>(Ob, WoT, bo, out);
}
